// SKipPool_Full_25890062861049
// MI455X (gfx1250) — matrix operand order repaired and checked by reading, not compiled, not run on hardware
//
#include <hip/hip_runtime.h>
#include <cstddef>
#include <cstdint>

// ---- static problem configuration (matches reference) ----
#define NGRAPH 64
#define NPG    512
#define DIM    128
#define KP     256     // top-k per graph
#define FSW    16      // window width
#define RESW   241     // rows per graph = KP - FSW + 1

typedef __attribute__((ext_vector_type(16))) __bf16 bf16x16;
typedef __attribute__((ext_vector_type(8)))  float  floatx8;
typedef __attribute__((ext_vector_type(4)))  unsigned v4u;
typedef __attribute__((ext_vector_type(8)))  int      v8i;
typedef __attribute__((ext_vector_type(4)))  int      v4i;

union Frag { bf16x16 v; uint4 u[2]; };

__device__ __forceinline__ unsigned short f2bf(float f) {
  unsigned u = __float_as_uint(f);
  unsigned r = u + 0x7FFFu + ((u >> 16) & 1u);   // round-to-nearest-even
  return (unsigned short)(r >> 16);
}

// -------------------------------------------------------------------------
// K1: scores = (x @ w + b) / ||w||      (wave-per-row GEMV, wave32)
// -------------------------------------------------------------------------
__global__ void scores_kernel(const float* __restrict__ x,
                              const float* __restrict__ w,
                              const float* __restrict__ bptr,
                              float* __restrict__ scores) {
  const int wave = threadIdx.x >> 5, lane = threadIdx.x & 31;
  const int row = blockIdx.x * 8 + wave;
  __builtin_prefetch(x + ((size_t)row + 8) * DIM + lane * 4, 0, 1);
  const float4 xv = *(const float4*)(x + (size_t)row * DIM + lane * 4);
  const float4 wv = *(const float4*)(w + lane * 4);
  float s  = xv.x * wv.x + xv.y * wv.y + xv.z * wv.z + xv.w * wv.w;
  float wn = wv.x * wv.x + wv.y * wv.y + wv.z * wv.z + wv.w * wv.w;
  #pragma unroll
  for (int off = 16; off > 0; off >>= 1) {
    s  += __shfl_xor(s,  off, 32);
    wn += __shfl_xor(wn, off, 32);
  }
  if (lane == 0) scores[row] = (s + bptr[0]) * rsqrtf(wn);
}

// -------------------------------------------------------------------------
// K2: per-graph descending bitonic sort of 512 scores -> ordered top-256 perm
// -------------------------------------------------------------------------
__global__ void topk_kernel(const float* __restrict__ scores,
                            int* __restrict__ perm) {
  __shared__ float key[NPG];
  __shared__ int   val[NPG];
  const int g = blockIdx.x, t = threadIdx.x;
  for (int i = t; i < NPG; i += 256) { key[i] = scores[g * NPG + i]; val[i] = i; }
  __syncthreads();
  for (int k = 2; k <= NPG; k <<= 1) {
    for (int j = k >> 1; j > 0; j >>= 1) {
      for (int i = t; i < NPG; i += 256) {
        const int ixj = i ^ j;
        if (ixj > i) {
          const bool up = ((i & k) == 0);
          const float a = key[i], c = key[ixj];
          const bool sw = up ? (a < c) : (a > c);   // final pass -> descending
          if (sw) {
            key[i] = c; key[ixj] = a;
            const int tv = val[i]; val[i] = val[ixj]; val[ixj] = tv;
          }
        }
      }
      __syncthreads();
    }
  }
  if (t < KP) perm[g * KP + t] = g * NPG + val[t];
}

// -------------------------------------------------------------------------
// K3: gather xp = x[perm] as bf16; fused norms + attention GEMVs
// -------------------------------------------------------------------------
__global__ void gather_kernel(const float* __restrict__ x,
                              const int* __restrict__ perm,
                              const float* __restrict__ att,
                              unsigned short* __restrict__ xp,
                              float* __restrict__ n2,
                              float* __restrict__ asrc,
                              float* __restrict__ adst) {
  const int wave = threadIdx.x >> 5, lane = threadIdx.x & 31;
  const int p = blockIdx.x * 8 + wave;
  const int row = perm[p];
  const float4 xv = *(const float4*)(x + (size_t)row * DIM + lane * 4);
  const float4 av = *(const float4*)(att + lane * 4);
  const float4 bv = *(const float4*)(att + DIM + lane * 4);
  float nn = xv.x * xv.x + xv.y * xv.y + xv.z * xv.z + xv.w * xv.w;
  float sa = xv.x * av.x + xv.y * av.y + xv.z * av.z + xv.w * av.w;
  float sb = xv.x * bv.x + xv.y * bv.y + xv.z * bv.z + xv.w * bv.w;
  #pragma unroll
  for (int off = 16; off > 0; off >>= 1) {
    nn += __shfl_xor(nn, off, 32);
    sa += __shfl_xor(sa, off, 32);
    sb += __shfl_xor(sb, off, 32);
  }
  uint2 pk;
  pk.x = (unsigned)f2bf(xv.x) | ((unsigned)f2bf(xv.y) << 16);
  pk.y = (unsigned)f2bf(xv.z) | ((unsigned)f2bf(xv.w) << 16);
  *(uint2*)(xp + (size_t)p * DIM + lane * 4) = pk;
  if (lane == 0) { n2[p] = nn; asrc[p] = sa; adst[p] = sb; }
}

// -------------------------------------------------------------------------
// K4: per-graph: TDM-stage tile -> WMMA Gram band -> gk; softmax; per-row
//     sort; column sums; gumbel-softmax threshold; relu.
// Shared layout (bytes): xp 65536 | n2 1024 | as 1024 | ad 1024 |
//                        gk 16384 | gksort 16384 | logits 64  -> 101440
// -------------------------------------------------------------------------
#define SH_XP 0
#define SH_N2 65536
#define SH_AS 66560
#define SH_AD 67584
#define SH_GK 68608
#define SH_GS 84992
#define SH_LG 101376
#define SH_TOTAL 101440

__global__ void graph_kernel(const unsigned short* __restrict__ xp_g,
                             const float* __restrict__ n2_g,
                             const float* __restrict__ as_g,
                             const float* __restrict__ ad_g,
                             const float* __restrict__ gu,
                             const int* __restrict__ epoch_p,
                             float* __restrict__ out) {
  extern __shared__ char smem[];
  unsigned short* shXP = (unsigned short*)(smem + SH_XP);
  float* shN2 = (float*)(smem + SH_N2);
  float* shAS = (float*)(smem + SH_AS);
  float* shAD = (float*)(smem + SH_AD);
  float* shGK = (float*)(smem + SH_GK);
  float* shGS = (float*)(smem + SH_GS);
  float* shLG = (float*)(smem + SH_LG);

  const int g = blockIdx.x, tid = threadIdx.x;
  const int wave = tid >> 5, lane = tid & 31;

  // ---- Phase A: Tensor Data Mover stages the 64KB bf16 tile into LDS ----
  if (wave == 0) {
    // D# group 0: count=1 | lds_addr[63:32] | global_addr[120:64] | type=2
    const unsigned long long ga =
        (unsigned long long)(uintptr_t)(xp_g + (size_t)g * KP * DIM);
    const unsigned lds_off = (unsigned)(uintptr_t)shXP;  // LDS byte offset
    v4u g0;
    g0[0] = 1u;                                          // count = 1 valid D#
    g0[1] = lds_off;
    g0[2] = (unsigned)(ga & 0xFFFFFFFFu);
    g0[3] = (unsigned)((ga >> 32) & 0x01FFFFFFu) | (2u << 30);  // type=2
    // D# group 1: 2D tensor of bf16: dims 128 x 256, tile 128 x 256, stride 128
    v8i g1;
    g1[0] = 0x00010000;          // workgroup_mask=0 | data_size=1 (2 bytes)
    g1[1] = (int)(128u << 16);   // atomic_barrier_addr=0 | tensor_dim0.lo16=128
    g1[2] = (int)(256u << 16);   // tensor_dim0.hi16=0 | tensor_dim1.lo16=256
    g1[3] = (int)(128u << 16);   // tensor_dim1.hi16=0 | tile_dim0=128
    g1[4] = 256;                 // tile_dim1=256 | tile_dim2=0
    g1[5] = 128;                 // tensor_dim0_stride = 128 (lo32)
    g1[6] = 0;                   // stride hi | tensor_dim1_stride.lo16
    g1[7] = 0;
    const v4i gz4 = {0, 0, 0, 0};                        // groups 2/3 unused (2D)
    const v8i gz8 = {0, 0, 0, 0, 0, 0, 0, 0};
    __builtin_amdgcn_tensor_load_to_lds(g0, g1, gz4, gz4, gz8, 0);
    __builtin_amdgcn_s_wait_tensorcnt((short)0);
  }
  // per-node scalars via plain loads (tiny)
  shN2[tid] = n2_g[g * KP + tid];
  shAS[tid] = as_g[g * KP + tid];
  shAD[tid] = ad_g[g * KP + tid];
  if (tid < FSW) shLG[tid] = 0.0f;
  __syncthreads();

  // ---- Phase B: banded Gram via v_wmma_f32_16x16x32_bf16 (branch-free) ----
  const int m = lane & 15, h = lane >> 4;   // A-row / B-col index, K-half index

  for (int blk = wave; blk < 16; blk += 8) {            // wave-uniform
    floatx8 acc0 = {0.f,0.f,0.f,0.f,0.f,0.f,0.f,0.f};
    floatx8 acc1 = {0.f,0.f,0.f,0.f,0.f,0.f,0.f,0.f};
    const int nb = (blk < 15) ? (blk + 1) * 16 : 0;     // clamped, uniform select
    const char* rowAp  = (const char*)shXP + (size_t)(blk * 16 + m) * (DIM * 2);
    const char* rowB1p = (const char*)shXP + (size_t)(nb + m) * (DIM * 2);
    #pragma unroll
    for (int kk = 0; kk < DIM; kk += 32) {
      Frag fa, fb0, fb1;
      // A (16x32 bf16): lane(m,h) holds row m, K = 16h..16h+15 (contiguous)
      fa.u[0] = *(const uint4*)(rowAp + (kk + 16 * h) * 2);
      fa.u[1] = *(const uint4*)(rowAp + (kk + 16 * h) * 2 + 16);
      // B (32x16 bf16): lane(n,h) holds col n, K = 16h..16h+15 (contiguous)
      fb0.u[0] = *(const uint4*)(rowAp + (kk + 16 * h) * 2);
      fb0.u[1] = *(const uint4*)(rowAp + (kk + 16 * h) * 2 + 16);
      fb1.u[0] = *(const uint4*)(rowB1p + (kk + 16 * h) * 2);
      fb1.u[1] = *(const uint4*)(rowB1p + (kk + 16 * h) * 2 + 16);
      acc0 = __builtin_amdgcn_wmma_f32_16x16x32_bf16(
          false, fa.v, false, fb0.v, (short)0, acc0, false, false);
      acc1 = __builtin_amdgcn_wmma_f32_16x16x32_bf16(
          false, fa.v, false, fb1.v, (short)0, acc1, false, false);
    }
    // C/D layout: VGPR t, lane n+16h -> element (M = t+8h, N = n)
    #pragma unroll
    for (int t = 0; t < 8; ++t) {
      const int M = t + 8 * h;
      const int r = blk * 16 + M;
      const int n = m;
      if (r < RESW) {
        {
          const int j = n - M;                          // diagonal tile: j >= 0
          if (j >= 0) {
            const int d = blk * 16 + n;
            const float d2 = shN2[r] + shN2[d] - 2.0f * acc0[t];
            const float dist = (j == 0) ? 0.0f : ((d2 > 0.0f) ? __fsqrt_rn(d2) : 0.0f);
            shGK[r * FSW + j] = __expf(-0.5f * dist) - 1e-5f;
          }
        }
        {
          const int j = 16 + n - M;                     // off-diag tile: j in 1..15
          if (blk < 15 && j < 16) {                     // store-only divergence
            const int d = (blk + 1) * 16 + n;
            const float d2 = shN2[r] + shN2[d] - 2.0f * acc1[t];
            const float dist = (d2 > 0.0f) ? __fsqrt_rn(d2) : 0.0f;
            shGK[r * FSW + j] = __expf(-0.5f * dist) - 1e-5f;
          }
        }
      }
    }
  }
  __syncthreads();

  // ---- Phase C: per-row softmax + sort-by-gk + column sums ----
  if (tid < RESW) {
    const int r = tid;
    float gk[FSW], sm[FSW];
    const float as = shAS[r];
    float mx = -1e30f;
    #pragma unroll
    for (int j = 0; j < FSW; ++j) {
      gk[j] = shGK[r * FSW + j];
      sm[j] = as + shAD[r + j];
      mx = fmaxf(mx, sm[j]);
    }
    float ssum = 0.0f;
    #pragma unroll
    for (int j = 0; j < FSW; ++j) { sm[j] = __expf(sm[j] - mx); ssum += sm[j]; }
    const float sinv = 1.0f / ssum;
    #pragma unroll
    for (int j = 0; j < FSW; ++j) sm[j] *= sinv;

    // fully unrolled 16-element bitonic sort, descending by gk, carrying sm
    #pragma unroll
    for (int k = 2; k <= FSW; k <<= 1) {
      #pragma unroll
      for (int jj = k >> 1; jj > 0; jj >>= 1) {
        #pragma unroll
        for (int i = 0; i < FSW; ++i) {
          const int ix = i ^ jj;
          if (ix > i) {
            const bool up = ((i & k) == 0);
            const bool sw = up ? (gk[i] < gk[ix]) : (gk[i] > gk[ix]);
            if (sw) {
              float tg = gk[i]; gk[i] = gk[ix]; gk[ix] = tg;
              float ts = sm[i]; sm[i] = sm[ix]; sm[ix] = ts;
            }
          }
        }
      }
    }
    #pragma unroll
    for (int j = 0; j < FSW; ++j) {
      shGS[r * FSW + j] = gk[j];
      atomicAdd(&shLG[j], sm[j]);                       // ds_add_f32
    }
  }
  __syncthreads();

  // ---- Phase D: gumbel softmax threshold + relu ----
  if (tid < RESW) {
    const int r = tid;
    const float ep = (float)(*epoch_p);
    const float tau0 = 10.0f * __powf(0.01f, ep * 0.01f);
    const float itau = 1.0f / tau0;
    const size_t e0 = ((size_t)g * RESW + r) * FSW;
    float z[FSW];
    float mx = -1e30f;
    #pragma unroll
    for (int j = 0; j < FSW; ++j) {
      const float lg = (j == 0) ? -1e30f : shLG[j];
      const float gb = -__logf(gu[e0 + j]);
      z[j] = (gb + lg) * itau;
      mx = fmaxf(mx, z[j]);
    }
    float zsum = 0.0f;
    #pragma unroll
    for (int j = 0; j < FSW; ++j) { z[j] = __expf(z[j] - mx); zsum += z[j]; }
    const float zinv = 1.0f / zsum;
    float ths = 0.0f;
    #pragma unroll
    for (int j = 0; j < FSW; ++j) ths += shGS[r * FSW + j] * z[j] * zinv;
    #pragma unroll
    for (int j = 0; j < FSW; ++j)
      out[e0 + j] = fmaxf(shGK[r * FSW + j] - ths, 0.0f);
  }
}

// -------------------------------------------------------------------------
extern "C" void kernel_launch(void* const* d_in, const int* in_sizes, int n_in,
                              void* d_out, int out_size, void* d_ws, size_t ws_size,
                              hipStream_t stream) {
  (void)in_sizes; (void)n_in; (void)out_size; (void)ws_size;
  const float* x       = (const float*)d_in[0];
  // d_in[1] edge_index, d_in[2] batch: unused by the reference computation
  const int*   epoch   = (const int*)d_in[3];
  const float* score_w = (const float*)d_in[4];
  const float* score_b = (const float*)d_in[5];
  const float* att     = (const float*)d_in[6];
  const float* gu      = (const float*)d_in[7];

  char* ws = (char*)d_ws;
  float*          scores = (float*)(ws + 0);               // 32768 * 4  = 131072
  int*            perm   = (int*)(ws + 131072);            // 16384 * 4  =  65536
  unsigned short* xp     = (unsigned short*)(ws + 196608); // 16384*128*2 = 4194304
  float*          n2     = (float*)(ws + 4390912);         // 16384 * 4
  float*          asrc   = (float*)(ws + 4456448);         // 16384 * 4
  float*          adst   = (float*)(ws + 4521984);         // 16384 * 4

  scores_kernel<<<(NGRAPH * NPG) / 8, 256, 0, stream>>>(x, score_w, score_b, scores);
  topk_kernel<<<NGRAPH, 256, 0, stream>>>(scores, perm);
  gather_kernel<<<(NGRAPH * KP) / 8, 256, 0, stream>>>(x, perm, att, xp, n2, asrc, adst);
  graph_kernel<<<NGRAPH, 256, SH_TOTAL, stream>>>(xp, n2, asrc, adst, gu, epoch,
                                                  (float*)d_out);
}
